// SparseAttention_86517821216553
// MI455X (gfx1250) — matrix operand order repaired and checked by reading, not compiled, not run on hardware
//
#include <hip/hip_runtime.h>
#include <hip/hip_bf16.h>

// ---------- problem constants ----------
#define LL 1024
#define BB 8
#define EE 512
#define HH 8
#define DD 64
#define MROWS (LL * BB)          // 8192 token rows for the projections
#define SCALE 0.125f             // 1/sqrt(64)
#define THRESH 0.1f

// ---------- CDNA5 WMMA types ----------
typedef __bf16 v16bf __attribute__((ext_vector_type(16)));
typedef float  v8f   __attribute__((ext_vector_type(8)));

union Frag {                      // 16 bf16 = 32B per lane = 8 VGPRs
    v16bf v;
    uint4 q[2];
};

__device__ __forceinline__ v8f wmma_bf16(const Frag& a, const Frag& b, v8f c) {
    // D = A(16x32 bf16) * B(32x16 bf16) + C(16x16 f32)
    return __builtin_amdgcn_wmma_f32_16x16x32_bf16(
        false, a.v, false, b.v, (short)0, c, false, false);
}

// native f32 -> bf16 convert (let the compiler use v_cvt_* on gfx1250)
__device__ __forceinline__ unsigned short f2bf(float f) {
    union { __bf16 b; unsigned short u; } cv;
    cv.b = (__bf16)f;
    return cv.u;
}

// convert 8 contiguous f32 -> 8 bf16, one b128 load pair + one b128 LDS store
__device__ __forceinline__ void stage8(const float* __restrict__ src,
                                       unsigned short* __restrict__ dst) {
    float4 f0 = *(const float4*)(src);
    float4 f1 = *(const float4*)(src + 4);
    union { unsigned short u[8]; uint4 q; } o;
    o.u[0] = f2bf(f0.x); o.u[1] = f2bf(f0.y);
    o.u[2] = f2bf(f0.z); o.u[3] = f2bf(f0.w);
    o.u[4] = f2bf(f1.x); o.u[5] = f2bf(f1.y);
    o.u[6] = f2bf(f1.z); o.u[7] = f2bf(f1.w);
    *(uint4*)dst = o.q;
}

__device__ __forceinline__ v8f zero8() {
    v8f z;
#pragma unroll
    for (int i = 0; i < 8; ++i) z[i] = 0.0f;
    return z;
}

// =====================================================================
// Kernel 1: Y = X @ W^T + bias, output bf16 (row-major 8192 x 512)
//   Block: 128 thr (4 waves), tile 128(M) x 64(N), K step 32.
//   Each wave: 32 rows -> 2 A-frags x 4 B-frags = 8 WMMA / K-step
//   (B-fragment reused across both M sub-tiles).
// =====================================================================
__global__ __launch_bounds__(128) void proj_kernel(
    const float* __restrict__ X, const float* __restrict__ W,
    const float* __restrict__ bias, unsigned short* __restrict__ Yb) {

    __shared__ __align__(16) unsigned short Asm[128][40]; // pad 8 -> 80B stride
    __shared__ __align__(16) unsigned short Bsm[64][40];

    const int tid  = threadIdx.x;
    const int wave = tid >> 5;
    const int lane = tid & 31;
    const int m0 = blockIdx.y * 128;
    const int n0 = blockIdx.x * 64;

    v8f c[2][4];
#pragma unroll
    for (int mi = 0; mi < 2; ++mi)
#pragma unroll
        for (int ns = 0; ns < 4; ++ns) c[mi][ns] = zero8();

    for (int k0 = 0; k0 < EE; k0 += 32) {
        __syncthreads();
        // stage A 128x32 (4096 elems, 8/thread/pass, 4 passes)
#pragma unroll
        for (int p = 0; p < 4; ++p) {
            int i = (tid + p * 128) * 8;
            int r = i >> 5, cc = i & 31;
            stage8(&X[(size_t)(m0 + r) * EE + k0 + cc], &Asm[r][cc]);
        }
        // stage B 64x32 (2048 elems, 2 passes)
#pragma unroll
        for (int p = 0; p < 2; ++p) {
            int i = (tid + p * 128) * 8;
            int r = i >> 5, cc = i & 31;
            stage8(&W[(size_t)(n0 + r) * EE + k0 + cc], &Bsm[r][cc]);
        }
        if (k0 + 32 < EE) {   // prefetch next K tile (global_prefetch_b8)
            __builtin_prefetch(&X[(size_t)(m0 + tid) * EE + k0 + 32], 0, 1);
            __builtin_prefetch(&W[(size_t)(n0 + (tid & 63)) * EE + k0 + 32], 0, 1);
        }
        __syncthreads();

        Frag a[2];
        const int ka = (lane >> 4) * 8;    // A: K chunks {ka..+7, ka+16..+23}
#pragma unroll
        for (int mi = 0; mi < 2; ++mi) {
            const int mr = wave * 32 + mi * 16 + (lane & 15);
            a[mi].q[0] = *(const uint4*)&Asm[mr][ka];
            a[mi].q[1] = *(const uint4*)&Asm[mr][ka + 16];
        }
#pragma unroll
        for (int ns = 0; ns < 4; ++ns) {
            Frag bf;
            const int nr  = ns * 16 + (lane & 15);
            const int kb0 = (lane >> 4) * 8;   // B: K chunks {kb0..+7, kb0+16..+23}
            bf.q[0] = *(const uint4*)&Bsm[nr][kb0];
            bf.q[1] = *(const uint4*)&Bsm[nr][kb0 + 16];
#pragma unroll
            for (int mi = 0; mi < 2; ++mi)
                c[mi][ns] = wmma_bf16(a[mi], bf, c[mi][ns]);
        }
    }

    // epilogue: + bias, f32 -> bf16, store
#pragma unroll
    for (int mi = 0; mi < 2; ++mi) {
        const int rbase = wave * 32 + mi * 16 + 8 * (lane >> 4);
#pragma unroll
        for (int ns = 0; ns < 4; ++ns) {
            const int col = n0 + ns * 16 + (lane & 15);
            const float bv = bias[col];
#pragma unroll
            for (int r = 0; r < 8; ++r) {
                Yb[(size_t)(m0 + rbase + r) * EE + col] = f2bf(c[mi][ns][r] + bv);
            }
        }
    }
}

// =====================================================================
// Kernel 2: fused scores + softmax + head-average + threshold + renorm
//   One block (256 thr, 8 waves) per (b, 16-row l-tile).
//   Q̂,K̂ are bf16 [row= l*B+b][E], K-contiguous per head -> fragments
//   load directly from global (NT GEMM, M=16, N=1024, K=64).
//   Dynamic LDS: sc[16][1024] f32 + acc[16][1024] f32 = 128 KB.
// =====================================================================
__global__ __launch_bounds__(256) void attn_kernel(
    const unsigned short* __restrict__ qb,
    const unsigned short* __restrict__ kb,
    float* __restrict__ sw) {

    extern __shared__ float smem[];
    float* sc  = smem;               // current head's 16 x 1024 scores/probs
    float* acc = smem + 16 * 1024;   // head-averaged probabilities

    const int tid  = threadIdx.x;
    const int wave = tid >> 5;
    const int lane = tid & 31;
    const int l0 = blockIdx.x * 16;
    const int b  = blockIdx.y;

    for (int h = 0; h < HH; ++h) {
        // ---- scores tile: rows l0..l0+15, cols wave*128..+127 ----
        Frag a0, a1;
        {
            const int m  = lane & 15;
            const int ka = (lane >> 4) * 8;
            const unsigned short* qrow =
                qb + ((size_t)(l0 + m) * BB + b) * EE + h * DD;
            a0.q[0] = *(const uint4*)(qrow + ka);
            a0.q[1] = *(const uint4*)(qrow + ka + 16);
            a1.q[0] = *(const uint4*)(qrow + 32 + ka);
            a1.q[1] = *(const uint4*)(qrow + 32 + ka + 16);
        }
#pragma unroll
        for (int ns = 0; ns < 8; ++ns) {
            const int s0 = wave * 128 + ns * 16;
            const unsigned short* krow =
                kb + ((size_t)(s0 + (lane & 15)) * BB + b) * EE + h * DD;
            Frag b0, b1;
            const int kb0 = (lane >> 4) * 8;
            b0.q[0] = *(const uint4*)(krow + kb0);
            b0.q[1] = *(const uint4*)(krow + kb0 + 16);
            b1.q[0] = *(const uint4*)(krow + 32 + kb0);
            b1.q[1] = *(const uint4*)(krow + 32 + kb0 + 16);

            v8f c = zero8();
            c = wmma_bf16(a0, b0, c);   // K = 0..31
            c = wmma_bf16(a1, b1, c);   // K = 32..63

            const int rbase = 8 * (lane >> 4);
            const int col   = s0 + (lane & 15);
#pragma unroll
            for (int r = 0; r < 8; ++r)
                sc[(rbase + r) * 1024 + col] = c[r] * SCALE;
        }
        __syncthreads();

        // ---- softmax per row (2 rows per wave), accumulate prob/H ----
#pragma unroll
        for (int rr = 0; rr < 2; ++rr) {
            const int row = wave * 2 + rr;
            float mx = -1e30f;
            for (int s = lane; s < 1024; s += 32)
                mx = fmaxf(mx, sc[row * 1024 + s]);
#pragma unroll
            for (int off = 16; off; off >>= 1)
                mx = fmaxf(mx, __shfl_xor(mx, off, 32));

            float sum = 0.0f;
            for (int s = lane; s < 1024; s += 32) {
                float e = __expf(sc[row * 1024 + s] - mx);
                sc[row * 1024 + s] = e;
                sum += e;
            }
#pragma unroll
            for (int off = 16; off; off >>= 1)
                sum += __shfl_xor(sum, off, 32);

            const float inv = (1.0f / sum) * (1.0f / HH);
            for (int s = lane; s < 1024; s += 32) {
                float p = sc[row * 1024 + s] * inv;
                if (h == 0) acc[row * 1024 + s] = p;
                else        acc[row * 1024 + s] += p;
            }
        }
        __syncthreads();
    }

    // ---- threshold + renormalize + emit sw [B,L,S] ----
#pragma unroll
    for (int rr = 0; rr < 2; ++rr) {
        const int row = wave * 2 + rr;
        float sum = 0.0f;
        for (int s = lane; s < 1024; s += 32) {
            float v = acc[row * 1024 + s];
            v = (v > THRESH) ? v : 0.0f;
            acc[row * 1024 + s] = v;
            sum += v;
        }
#pragma unroll
        for (int off = 16; off; off >>= 1)
            sum += __shfl_xor(sum, off, 32);

        const float inv = 1.0f / (sum + 1e-6f);
        float* dst = sw + ((size_t)b * LL + (l0 + row)) * LL;
        for (int s = lane; s < 1024; s += 32)
            dst[s] = acc[row * 1024 + s] * inv;
    }
}

// =====================================================================
// Kernel 3: out[l,b,e] = sum_s sw[b,l,s] * value[s,b,e]
//   Per b: GEMM M=1024(l) N=512(e) K=1024(s). sw staged f32->bf16; V
//   staged transposed (column e of V_b is strided) into LDS as bf16.
//   Block 128 thr, tile 128x64, K step 32, 8 WMMA / wave / K-step.
// =====================================================================
__global__ __launch_bounds__(128) void out_gemm_kernel(
    const float* __restrict__ sw, const float* __restrict__ value,
    float* __restrict__ out) {

    __shared__ __align__(16) unsigned short Asm[128][40]; // sw tile  [m][k]
    __shared__ __align__(16) unsigned short Bsm[64][40];  // V^T tile [e][k]

    const int tid  = threadIdx.x;
    const int wave = tid >> 5;
    const int lane = tid & 31;
    const int e0 = blockIdx.x * 64;
    const int l0 = blockIdx.y * 128;
    const int b  = blockIdx.z;

    v8f c[2][4];
#pragma unroll
    for (int mi = 0; mi < 2; ++mi)
#pragma unroll
        for (int ns = 0; ns < 4; ++ns) c[mi][ns] = zero8();

    for (int k0 = 0; k0 < LL; k0 += 32) {
        __syncthreads();
        // A: sw rows, 128x32 (vectorized, 4 passes)
#pragma unroll
        for (int p = 0; p < 4; ++p) {
            int i = (tid + p * 128) * 8;
            int r = i >> 5, cc = i & 31;
            stage8(&sw[((size_t)b * LL + l0 + r) * LL + k0 + cc], &Asm[r][cc]);
        }
        // B: V transposed. Each thread: float4 over e (coalesced), 4 scalar
        // bf16 stores into 4 LDS rows. 2048 elems / 4 = 512 slots, 4 passes.
#pragma unroll
        for (int p = 0; p < 4; ++p) {
            int i  = tid + p * 128;         // 0..511
            int kk = i >> 4;                // 0..31
            int ee = (i & 15) * 4;          // 0..60
            float4 f = *(const float4*)
                &value[((size_t)(k0 + kk) * BB + b) * EE + e0 + ee];
            Bsm[ee + 0][kk] = f2bf(f.x);
            Bsm[ee + 1][kk] = f2bf(f.y);
            Bsm[ee + 2][kk] = f2bf(f.z);
            Bsm[ee + 3][kk] = f2bf(f.w);
        }
        if (k0 + 32 < LL) {   // prefetch next K tile
            __builtin_prefetch(&sw[((size_t)b * LL + l0 + tid) * LL + k0 + 32], 0, 1);
            __builtin_prefetch(&value[((size_t)(k0 + 32 + (tid & 31)) * BB + b) * EE + e0], 0, 1);
        }
        __syncthreads();

        Frag a[2];
        const int ka = (lane >> 4) * 8;
#pragma unroll
        for (int mi = 0; mi < 2; ++mi) {
            const int mr = wave * 32 + mi * 16 + (lane & 15);
            a[mi].q[0] = *(const uint4*)&Asm[mr][ka];
            a[mi].q[1] = *(const uint4*)&Asm[mr][ka + 16];
        }
#pragma unroll
        for (int ns = 0; ns < 4; ++ns) {
            Frag bf;
            const int nr  = ns * 16 + (lane & 15);
            const int kb0 = (lane >> 4) * 8;
            bf.q[0] = *(const uint4*)&Bsm[nr][kb0];
            bf.q[1] = *(const uint4*)&Bsm[nr][kb0 + 16];
#pragma unroll
            for (int mi = 0; mi < 2; ++mi)
                c[mi][ns] = wmma_bf16(a[mi], bf, c[mi][ns]);
        }
    }

#pragma unroll
    for (int mi = 0; mi < 2; ++mi) {
        const int rbase = wave * 32 + mi * 16 + 8 * (lane >> 4);
#pragma unroll
        for (int ns = 0; ns < 4; ++ns) {
            const int col = e0 + ns * 16 + (lane & 15);
#pragma unroll
            for (int r = 0; r < 8; ++r) {
                out[(size_t)(l0 + rbase + r) * (BB * EE) + (size_t)b * EE + col] =
                    c[mi][ns][r];
            }
        }
    }
}

// =====================================================================
extern "C" void kernel_launch(void* const* d_in, const int* in_sizes, int n_in,
                              void* d_out, int out_size, void* d_ws, size_t ws_size,
                              hipStream_t stream) {
    (void)in_sizes; (void)n_in; (void)out_size; (void)ws_size;

    const float* query = (const float*)d_in[0];  // (L,B,E)
    const float* key   = (const float*)d_in[1];  // (L,B,E)
    const float* value = (const float*)d_in[2];  // (L,B,E)
    const float* Wq    = (const float*)d_in[3];  // (E,E)
    const float* bq    = (const float*)d_in[4];  // (E,)
    const float* Wk    = (const float*)d_in[5];  // (E,E)
    const float* bk    = (const float*)d_in[6];  // (E,)

    float* out = (float*)d_out;                       // (L,B,E)
    float* sw  = out + (size_t)LL * BB * EE;          // (B,L,S)

    unsigned short* qb = (unsigned short*)d_ws;                 // bf16 Q̂ 8 MB
    unsigned short* kb = qb + (size_t)MROWS * EE;               // bf16 K̂ 8 MB

    // K1: projections (run twice: Q then K)
    dim3 g1(EE / 64, MROWS / 128);                    // (8, 64)
    proj_kernel<<<g1, 128, 0, stream>>>(query, Wq, bq, qb);
    proj_kernel<<<g1, 128, 0, stream>>>(key,   Wk, bk, kb);

    // K2: fused scores / softmax / head-avg / threshold -> sw
    size_t smem2 = 2 * 16 * 1024 * sizeof(float);     // 128 KB dynamic LDS
    attn_kernel<<<dim3(LL / 16, BB), 256, smem2, stream>>>(qb, kb, sw);

    // K3: out = sw @ V
    out_gemm_kernel<<<dim3(EE / 64, LL / 128, BB), 128, 0, stream>>>(sw, value, out);
}
